// SelfAttn1d_3582002725319
// MI455X (gfx1250) — matrix operand order repaired and checked by reading, not compiled, not run on hardware
//
#include <hip/hip_runtime.h>

typedef _Float16 f16;
typedef __attribute__((ext_vector_type(16))) _Float16 v16h;
typedef __attribute__((ext_vector_type(8)))  _Float16 v8h;
typedef __attribute__((ext_vector_type(4)))  _Float16 v4h;
typedef __attribute__((ext_vector_type(8)))  float    v8f;
typedef __attribute__((ext_vector_type(4)))  float    v4f;
typedef __attribute__((ext_vector_type(4)))  unsigned int v4u;
typedef __attribute__((ext_vector_type(8)))  int      v8i;
typedef __attribute__((ext_vector_type(4)))  int      v4i;

#define B_   8
#define N_   1024
#define F_   1024
#define H_   16
#define D_   64
#define G_   3072              // 3F
#define M_   (B_ * N_)         // 8192
#define EPSV 1e-5f
#define SCALE_QK 0.125f        // 1/sqrt(64)
#define NEGBIG (-1.0e9f)

// GEMM tiling
#define BK    32               // K-chunk per stage
#define LDT   40               // LDS row pitch in f16 (32 + 8 pad)
#define AS_BYTES (128 * LDT * 2)            // 10240 B (A tile, 128 rows)
#define WS_BYTES (128 * LDT * 2)            // 10240 B (W tile, 128 rows)
#define STAGE_BYTES (AS_BYTES + WS_BYTES)   // 20480 B
#define SMEM_GEMM (2 * STAGE_BYTES)         // 40960 B (ping-pong)

// TDM availability (device pass only; host pass parses fallback path)
#if defined(__HIP_DEVICE_COMPILE__) && \
    __has_builtin(__builtin_amdgcn_tensor_load_to_lds) && \
    __has_builtin(__builtin_amdgcn_s_wait_tensorcnt)
#define HAS_TDM 1
#else
#define HAS_TDM 0
#endif

#if HAS_TDM
// ---------------------------------------------------------------------------
// TDM 2D tile load: global (row-major, stride_e elements/row, 2B elements)
// -> LDS at lds_off, rows padded 64B->80B (pad_interval=8x8B, pad_amount=4 DW)
// Descriptor packing per CDNA5 ISA section 8 (D# groups 0/1; groups 2/3 zero).
// ---------------------------------------------------------------------------
__device__ __forceinline__ void tdm_load_2d(unsigned lds_off, const void* gsrc,
                                            unsigned tensor_d0, unsigned tensor_d1,
                                            unsigned tile_d0, unsigned tile_d1,
                                            unsigned stride_e) {
  unsigned long long ga = (unsigned long long)gsrc;
  v4u g0 = {1u,                                    // count=1 (valid user D#)
            lds_off,                               // lds_addr (bytes)
            (unsigned)ga,                          // global_addr[31:0]
            (unsigned)((ga >> 32) & 0x01ffffffULL) // global_addr[56:32]
                | (2u << 30)};                     // type=2 (image)
  v8i g1;
  g1[0] = (int)((1u << 16)    // data_size = 1 -> 2 bytes
              | (1u << 20)    // pad_enable
              | (3u << 22)    // pad_interval: 1<<3 8B-units = 64B
              | (3u << 25));  // pad_amount: 4 DWORDs = 16B
  g1[1] = (int)((tensor_d0 & 0xffffu) << 16);
  g1[2] = (int)(((tensor_d0 >> 16) & 0xffffu) | ((tensor_d1 & 0xffffu) << 16));
  g1[3] = (int)(((tensor_d1 >> 16) & 0xffffu) | ((tile_d0 & 0xffffu) << 16));
  g1[4] = (int)(tile_d1 & 0xffffu);   // tile_dim1; tile_dim2 = 0
  g1[5] = (int)stride_e;              // tensor_dim0_stride[31:0] (elements)
  g1[6] = 0;
  g1[7] = 0;
  v4i z4 = {0, 0, 0, 0};
#if __clang_major__ >= 23
  v8i z8 = {0, 0, 0, 0, 0, 0, 0, 0};
  __builtin_amdgcn_tensor_load_to_lds(g0, g1, z4, z4, z8, 0);
#else
  __builtin_amdgcn_tensor_load_to_lds(g0, g1, z4, z4, 0);
#endif
}
#endif  // HAS_TDM

// ---------------------------------------------------------------------------
// Kernel 1: f32 -> f16 conversion (weights)
// ---------------------------------------------------------------------------
__global__ __launch_bounds__(256) void cvt_f32_f16_kernel(
    const float* __restrict__ src, f16* __restrict__ dst, int n) {
  int i = (blockIdx.x * 256 + threadIdx.x) * 4;
  if (i < n) {
    v4f s = *(const v4f*)(src + i);
    v4h d;
    d.x = (f16)s.x; d.y = (f16)s.y; d.z = (f16)s.z; d.w = (f16)s.w;
    *(v4h*)(dst + i) = d;
  }
}

// ---------------------------------------------------------------------------
// Kernel 2: LayerNorm per row (F=1024, block=256, each thread 4 elems)
// ---------------------------------------------------------------------------
__global__ __launch_bounds__(256) void layernorm_kernel(
    const float* __restrict__ x, const float* __restrict__ lw,
    const float* __restrict__ lb, f16* __restrict__ out) {
  const int row = blockIdx.x;
  const int t = threadIdx.x;
  const float* xr = x + (size_t)row * F_;
  v4f v = *(const v4f*)(xr + t * 4);
  float s = v.x + v.y + v.z + v.w;
#pragma unroll
  for (int m = 1; m < 32; m <<= 1) s += __shfl_xor(s, m, 32);
  __shared__ float red[8];
  const int wv = t >> 5, ln = t & 31;
  if (ln == 0) red[wv] = s;
  __syncthreads();
  float tot = red[0] + red[1] + red[2] + red[3] + red[4] + red[5] + red[6] + red[7];
  const float mu = tot * (1.0f / F_);
  const float d0 = v.x - mu, d1 = v.y - mu, d2 = v.z - mu, d3 = v.w - mu;
  float q = d0 * d0 + d1 * d1 + d2 * d2 + d3 * d3;
#pragma unroll
  for (int m = 1; m < 32; m <<= 1) q += __shfl_xor(q, m, 32);
  __syncthreads();
  if (ln == 0) red[wv] = q;
  __syncthreads();
  float qt = red[0] + red[1] + red[2] + red[3] + red[4] + red[5] + red[6] + red[7];
  const float rstd = rsqrtf(qt * (1.0f / F_) + EPSV);
  v4f w4 = *(const v4f*)(lw + t * 4);
  v4f b4 = *(const v4f*)(lb + t * 4);
  v4h o;
  o.x = (f16)(d0 * rstd * w4.x + b4.x);
  o.y = (f16)(d1 * rstd * w4.y + b4.y);
  o.z = (f16)(d2 * rstd * w4.z + b4.z);
  o.w = (f16)(d3 * rstd * w4.w + b4.w);
  *(v4h*)(out + (size_t)row * F_ + t * 4) = o;
}

// ---------------------------------------------------------------------------
// Kernels 3/5: WMMA GEMM  C[M,Nc] = A[M,K] * W[Nc,K]^T  (+ epilogue)
// Block: 256 thr (8 waves), 128x128 block tile, 32x64 per wave (2x4 WMMA).
// Tiles staged via TDM DMA (ping-pong) when available, else manual loads.
// EPI==0: C -> f16 with +bias (QKV).  EPI==1: out = mask(x + C + bias) f32.
// ---------------------------------------------------------------------------
template <int EPI>
__global__ __launch_bounds__(256) void gemm_nt_kernel(
    const f16* __restrict__ A, const f16* __restrict__ W,
    const float* __restrict__ bias, int Ncols, int K,
    f16* __restrict__ Cf16, const float* __restrict__ Xres,
    const int* __restrict__ seq_lens, float* __restrict__ Cf32) {
  extern __shared__ char dynsmem[];
  const int m0 = blockIdx.x * 128;
  const int n0 = blockIdx.y * 128;
  const int t = threadIdx.x;
  const int wave = t >> 5;
  const int lane = t & 31;
  const int half = lane >> 4;
  const int lr = lane & 15;
  const int wm = (wave & 3) * 32;   // 4 M positions
  const int wn = (wave >> 2) * 64;  // 2 N positions

  v8f acc[2][4] = {};

  // one K-chunk of compute against a staged LDS buffer
  auto compute_stage = [&](const f16* As, const f16* Ws) {
    v16h af[2], bf[4];
#pragma unroll
    for (int i = 0; i < 2; ++i) {
      union { v16h v; v8h h[2]; } u;
      const f16* ap = As + (wm + i * 16 + lr) * LDT;
      u.h[0] = *(const v8h*)(ap + half * 8);
      u.h[1] = *(const v8h*)(ap + 16 + half * 8);
      af[i] = u.v;
    }
#pragma unroll
    for (int j = 0; j < 4; ++j) {
      union { v16h v; v8h h[2]; } u;
      const f16* bp = Ws + (wn + j * 16 + lr) * LDT + half * 8;
      u.h[0] = *(const v8h*)(bp);
      u.h[1] = *(const v8h*)(bp + 16);
      bf[j] = u.v;
    }
#pragma unroll
    for (int i = 0; i < 2; ++i)
#pragma unroll
      for (int j = 0; j < 4; ++j)
        acc[i][j] = __builtin_amdgcn_wmma_f32_16x16x32_f16(
            false, af[i], false, bf[j], (short)0, acc[i][j], false, false);
  };

  const int nstage = K / BK;

#if HAS_TDM
  // ---- TDM ping-pong pipeline: wave 0 issues DMA, TENSORcnt + barrier sync.
  if (wave == 0) {
    tdm_load_2d(0u, A + (size_t)m0 * K, (unsigned)K, 128u, BK, 128u, (unsigned)K);
    tdm_load_2d((unsigned)AS_BYTES, W + (size_t)n0 * K, (unsigned)K, 128u, BK,
                128u, (unsigned)K);
  }
  for (int s = 0; s < nstage; ++s) {
    if (wave == 0) __builtin_amdgcn_s_wait_tensorcnt(0);
    __syncthreads();  // stage s data visible; buf[s^1] readers (stage s-1) done
    if (wave == 0 && (s + 1) < nstage) {
      const unsigned buf = (unsigned)((s + 1) & 1) * STAGE_BYTES;
      const int kk = (s + 1) * BK;
      tdm_load_2d(buf, A + (size_t)m0 * K + kk, (unsigned)K, 128u, BK, 128u,
                  (unsigned)K);
      tdm_load_2d(buf + AS_BYTES, W + (size_t)n0 * K + kk, (unsigned)K, 128u,
                  BK, 128u, (unsigned)K);
    }
    const f16* As = (const f16*)(dynsmem + (size_t)(s & 1) * STAGE_BYTES);
    compute_stage(As, As + 128 * LDT);
  }
#else
  // ---- fallback: manual global->LDS staging, single buffer
  f16* As = (f16*)dynsmem;
  f16* Ws = As + 128 * LDT;
  const int arow = t >> 2;        // 0..63
  const int ac = (t & 3) * 8;     // k-chunk of 8 f16
  for (int s = 0; s < nstage; ++s) {
    const int kk = s * BK;
    __syncthreads();
#pragma unroll
    for (int p = 0; p < 2; ++p) {
      const int r = p * 64 + arow;
      *(v8h*)(As + r * LDT + ac) = *(const v8h*)(A + (size_t)(m0 + r) * K + kk + ac);
      *(v8h*)(Ws + r * LDT + ac) = *(const v8h*)(W + (size_t)(n0 + r) * K + kk + ac);
    }
    __syncthreads();
    compute_stage(As, Ws);
  }
#endif

  // Epilogue: C layout -> lane (half) holds rows (half*8 + r), col = lr
#pragma unroll
  for (int i = 0; i < 2; ++i)
#pragma unroll
    for (int j = 0; j < 4; ++j) {
      const int col = n0 + wn + j * 16 + lr;
      const float bv = bias[col];
#pragma unroll
      for (int r = 0; r < 8; ++r) {
        const int row = m0 + wm + i * 16 + half * 8 + r;
        const float val = acc[i][j][r] + bv;
        if (EPI == 0) {
          Cf16[(size_t)row * Ncols + col] = (f16)val;
        } else {
          const int bidx = row >> 10;      // / N
          const int nidx = row & (N_ - 1);
          const bool ok = nidx < seq_lens[bidx];
          const float xv = Xres[(size_t)row * Ncols + col];
          Cf32[(size_t)row * Ncols + col] = ok ? (xv + val) : 0.0f;
        }
      }
    }
}

// ---------------------------------------------------------------------------
// Kernel 4: flash attention.  Block = 256 thr (8 waves) handles one (b,h)
// and 128 q rows (16 per wave).  32 keys per iteration, online softmax.
// ---------------------------------------------------------------------------
__global__ __launch_bounds__(256) void attention_kernel(
    const f16* __restrict__ qkv, const int* __restrict__ seq_lens,
    f16* __restrict__ wvout) {
  const int qblk = blockIdx.x;            // 0..7
  const int h = blockIdx.y;               // 0..15
  const int b = blockIdx.z;               // 0..7
  const int t = threadIdx.x;
  const int wave = t >> 5, lane = t & 31;
  const int half = lane >> 4, lr = lane & 15;
  const int slen = seq_lens[b];

  constexpr int LK = 72;                  // K-tile pitch (64+8)
  constexpr int LV = 40;                  // Vt pitch (32+8)
  constexpr int LP = 40;                  // P pitch (32+8)
  __shared__ f16 Ks[32 * LK];             // 32 keys x 64 d
  __shared__ f16 Vt[64 * LV];             // transposed: 64 d x 32 keys
  __shared__ f16 Ps[8][16 * LP];          // per-wave P scratch

  const size_t base = (size_t)b * N_ * G_;
  const int q0 = qblk * 128 + wave * 16;

  // Q fragments (A layout), rows q0+lr, two 32-d chunks
  v16h aq[2];
#pragma unroll
  for (int s = 0; s < 2; ++s) {
    union { v16h v; v8h h2[2]; } u;
    const f16* qr = qkv + base + (size_t)(q0 + lr) * G_ + h * D_;
    u.h2[0] = *(const v8h*)(qr + s * 32 + half * 8);
    u.h2[1] = *(const v8h*)(qr + s * 32 + 16 + half * 8);
    aq[s] = u.v;
  }

  float mrow[8], lrow[8];
  v8f o[4] = {};
#pragma unroll
  for (int r = 0; r < 8; ++r) { mrow[r] = -3.0e38f; lrow[r] = 0.0f; }

  const int krow = t >> 3;                // 0..31
  const int kc = (t & 7) * 8;             // 0..56

  for (int kb = 0; kb < N_; kb += 32) {
    __syncthreads();
    // Stage K tile raw, V tile transposed
    {
      const f16* ksrc = qkv + base + (size_t)(kb + krow) * G_ + F_ + h * D_ + kc;
      *(v8h*)(Ks + krow * LK + kc) = *(const v8h*)ksrc;
      const f16* vsrc = qkv + base + (size_t)(kb + krow) * G_ + 2 * F_ + h * D_ + kc;
      v8h vd = *(const v8h*)vsrc;
#pragma unroll
      for (int e = 0; e < 8; ++e) Vt[(kc + e) * LV + krow] = vd[e];
    }
    __syncthreads();

    // S = Q * K^T : two 16x16 key tiles, K-dim = d (2 WMMA steps each)
    v8f s0 = {}, s1 = {};
#pragma unroll
    for (int s = 0; s < 2; ++s) {
      union { v16h v; v8h h2[2]; } b0, b1;
      const f16* kp0 = Ks + lr * LK + s * 32 + half * 8;
      b0.h2[0] = *(const v8h*)(kp0);
      b0.h2[1] = *(const v8h*)(kp0 + 16);
      const f16* kp1 = Ks + (16 + lr) * LK + s * 32 + half * 8;
      b1.h2[0] = *(const v8h*)(kp1);
      b1.h2[1] = *(const v8h*)(kp1 + 16);
      s0 = __builtin_amdgcn_wmma_f32_16x16x32_f16(false, aq[s], false, b0.v,
                                                  (short)0, s0, false, false);
      s1 = __builtin_amdgcn_wmma_f32_16x16x32_f16(false, aq[s], false, b1.v,
                                                  (short)0, s1, false, false);
    }
    // scale + key-validity mask (column key index = kb + tile*16 + lr)
    const bool ok0 = (kb + lr) < slen;
    const bool ok1 = (kb + 16 + lr) < slen;
    float alpha[8];
#pragma unroll
    for (int r = 0; r < 8; ++r) {
      const float e0 = ok0 ? s0[r] * SCALE_QK : NEGBIG;
      const float e1 = ok1 ? s1[r] * SCALE_QK : NEGBIG;
      float mx = fmaxf(e0, e1);
      mx = fmaxf(mx, __shfl_xor(mx, 1, 32));
      mx = fmaxf(mx, __shfl_xor(mx, 2, 32));
      mx = fmaxf(mx, __shfl_xor(mx, 4, 32));
      mx = fmaxf(mx, __shfl_xor(mx, 8, 32));
      const float mnew = fmaxf(mrow[r], mx);
      const float p0 = __expf(e0 - mnew);
      const float p1 = __expf(e1 - mnew);
      float rs = p0 + p1;
      rs += __shfl_xor(rs, 1, 32);
      rs += __shfl_xor(rs, 2, 32);
      rs += __shfl_xor(rs, 4, 32);
      rs += __shfl_xor(rs, 8, 32);
      alpha[r] = __expf(mrow[r] - mnew);
      lrow[r] = lrow[r] * alpha[r] + rs;
      mrow[r] = mnew;
      // write P (C layout -> LDS, row-major 16x32) for A-layout reload
      f16* pr = &Ps[wave][(half * 8 + r) * LP];
      pr[lr] = (f16)p0;
      pr[16 + lr] = (f16)p1;
    }
    // rescale accumulator
#pragma unroll
    for (int tt = 0; tt < 4; ++tt)
#pragma unroll
      for (int r = 0; r < 8; ++r) o[tt][r] *= alpha[r];

    asm volatile("s_wait_dscnt 0" ::: "memory");

    // reload P as A fragment: row = lr, k-chunks {half*8, 16+half*8}
    union { v16h v; v8h h2[2]; } ap;
    ap.h2[0] = *(const v8h*)(&Ps[wave][lr * LP + half * 8]);
    ap.h2[1] = *(const v8h*)(&Ps[wave][lr * LP + 16 + half * 8]);

    // O += P * V  (four 16-d tiles)
#pragma unroll
    for (int tt = 0; tt < 4; ++tt) {
      union { v16h v; v8h h2[2]; } bv;
      const f16* vp = Vt + (tt * 16 + lr) * LV + half * 8;
      bv.h2[0] = *(const v8h*)(vp);
      bv.h2[1] = *(const v8h*)(vp + 16);
      o[tt] = __builtin_amdgcn_wmma_f32_16x16x32_f16(false, ap.v, false, bv.v,
                                                     (short)0, o[tt], false, false);
    }
  }

  // finalize: wv[b, q, h*64 + d]
#pragma unroll
  for (int r = 0; r < 8; ++r) {
    const float inv = 1.0f / lrow[r];
    const int q = q0 + half * 8 + r;
    f16* dst = wvout + ((size_t)(b * N_ + q)) * F_ + h * D_;
#pragma unroll
    for (int tt = 0; tt < 4; ++tt) dst[tt * 16 + lr] = (f16)(o[tt][r] * inv);
  }
}

// ---------------------------------------------------------------------------
extern "C" void kernel_launch(void* const* d_in, const int* in_sizes, int n_in,
                              void* d_out, int out_size, void* d_ws,
                              size_t ws_size, hipStream_t stream) {
  (void)in_sizes; (void)n_in; (void)out_size; (void)ws_size;
  const float* x     = (const float*)d_in[0];
  const int*   slen  = (const int*)d_in[1];
  const float* ln_w  = (const float*)d_in[2];
  const float* ln_b  = (const float*)d_in[3];
  const float* in_w  = (const float*)d_in[4];
  const float* in_b  = (const float*)d_in[5];
  const float* out_w = (const float*)d_in[6];
  const float* out_b = (const float*)d_in[7];
  float* out = (float*)d_out;

  char* ws = (char*)d_ws;
  size_t off = 0;
  auto alloc = [&](size_t bytes) -> void* {
    void* p = ws + off;
    off = (off + bytes + 255) & ~(size_t)255;
    return p;
  };
  f16* normA   = (f16*)alloc((size_t)M_ * F_ * sizeof(f16));
  f16* qkv     = (f16*)alloc((size_t)M_ * G_ * sizeof(f16));
  f16* wv      = (f16*)alloc((size_t)M_ * F_ * sizeof(f16));
  f16* in_w16  = (f16*)alloc((size_t)G_ * F_ * sizeof(f16));
  f16* out_w16 = (f16*)alloc((size_t)F_ * F_ * sizeof(f16));

  // 1) weights -> f16
  cvt_f32_f16_kernel<<<(G_ * F_ / 4 + 255) / 256, 256, 0, stream>>>(in_w, in_w16, G_ * F_);
  cvt_f32_f16_kernel<<<(F_ * F_ / 4 + 255) / 256, 256, 0, stream>>>(out_w, out_w16, F_ * F_);
  // 2) LayerNorm
  layernorm_kernel<<<M_, 256, 0, stream>>>(x, ln_w, ln_b, normA);
  // 3) QKV projection (8192 x 3072 x 1024)
  dim3 g1(M_ / 128, G_ / 128);
  gemm_nt_kernel<0><<<g1, 256, SMEM_GEMM, stream>>>(normA, in_w16, in_b, G_, F_,
                                                    qkv, nullptr, nullptr, nullptr);
  // 4) attention
  dim3 g2(N_ / 128, H_, B_);
  attention_kernel<<<g2, 256, 0, stream>>>(qkv, slen, wv);
  // 5) out projection + residual + mask (8192 x 1024 x 1024)
  dim3 g3(M_ / 128, F_ / 128);
  gemm_nt_kernel<1><<<g3, 256, SMEM_GEMM, stream>>>(wv, out_w16, out_b, F_, F_,
                                                    nullptr, x, slen, out);
}
